// Fraser_23098334117906
// MI455X (gfx1250) — matrix operand order repaired and checked by reading, not compiled, not run on hardware
//
#include <hip/hip_runtime.h>

typedef __bf16 bf16_t;
typedef __attribute__((ext_vector_type(16))) __bf16 v16bf;
typedef __attribute__((ext_vector_type(8)))  __bf16 v8bf;
typedef __attribute__((ext_vector_type(8)))  float  v8f;
typedef __attribute__((ext_vector_type(4)))  int    v4i;

#define EMB   128
#define NPOS  121
#define NROW  128
#define NLAY  5
#define NTAP  9
#define NSTAGE (NLAY * NTAP)
#define WSTRIDE 136              // padded row (bf16 elems): 272B = 68 DWORDs -> conflict-free b128
#define BOARD_H 20
#define BOARD_W 10

// ---- CDNA5 async global->LDS path (guarded; falls back to sync copy) ----
#if defined(__gfx1250__) && __has_builtin(__builtin_amdgcn_global_load_async_to_lds_b128)
#define HAVE_ASYNC 1
typedef __attribute__((address_space(1))) v4i glob_v4i;
typedef __attribute__((address_space(3))) v4i lds_v4i;
__device__ __forceinline__ void async_cp16(void* lds_dst, const void* g_src) {
    __builtin_amdgcn_global_load_async_to_lds_b128(
        (glob_v4i*)g_src, (lds_v4i*)lds_dst, 0, 0);
}
__device__ __forceinline__ void wait_async0() {
#if __has_builtin(__builtin_amdgcn_s_wait_asynccnt)
    __builtin_amdgcn_s_wait_asynccnt(0);
#else
    asm volatile("s_wait_asynccnt 0x0" ::: "memory");
#endif
}
#else
#define HAVE_ASYNC 0
#endif

__device__ __forceinline__ v16bf ld_frag32(const bf16_t* p) {
    // 32B = two 16B LDS loads, combined into one 16-elem bf16 fragment
    v8bf lo = *(const v8bf*)(p);
    v8bf hi = *(const v8bf*)(p + 8);
    v16bf r;
    #pragma unroll
    for (int i = 0; i < 8; ++i) { r[i] = lo[i]; r[i + 8] = hi[i]; }
    return r;
}

__device__ __forceinline__ v16bf ld_fragA(const bf16_t* p) {
    // A layout: per-lane K {0..7} at p, K {8..15} at p+8
    v8bf lo = *(const v8bf*)(p);
    v8bf hi = *(const v8bf*)(p + 8);
    v16bf r;
    #pragma unroll
    for (int i = 0; i < 8; ++i) { r[i] = lo[i]; r[i + 8] = hi[i]; }
    return r;
}

// ---------------------------------------------------------------------------
// Prep kernel: conv_w [l][ky][kx][ci][co] f32  ->  wT [l][tap][co][ci] bf16
// ---------------------------------------------------------------------------
__global__ void Fraser_prep_weights(const float* __restrict__ conv_w,
                                    bf16_t* __restrict__ wT) {
    int idx = blockIdx.x * 256 + threadIdx.x;       // over 5*9*128*128
    if (idx >= NLAY * NTAP * EMB * EMB) return;
    int co = idx % EMB;
    int ci = (idx / EMB) % EMB;
    int t  = (idx / (EMB * EMB)) % NTAP;
    int l  =  idx / (EMB * EMB * NTAP);
    float v = conv_w[idx];
    wT[((l * NTAP + t) * EMB + co) * EMB + ci] = (bf16_t)v;
}

// ---------------------------------------------------------------------------
// Main fused kernel: one workgroup (8 wave32) per sample.
// ---------------------------------------------------------------------------
__global__ __launch_bounds__(256) void Fraser_tetris_net(
    const int*   __restrict__ x,
    const float* __restrict__ embed_w,
    const bf16_t* __restrict__ wT,       // [45][128][128] bf16, co-major, ci inner
    const float* __restrict__ conv_b,
    const float* __restrict__ ln_scale,
    const float* __restrict__ ln_bias,
    const float* __restrict__ head_w,
    const float* __restrict__ head_b,
    float*       __restrict__ out)
{
    __shared__ __align__(16) bf16_t hA[NROW * WSTRIDE];
    __shared__ __align__(16) bf16_t hB[NROW * WSTRIDE];
    __shared__ __align__(16) bf16_t wS[2][NROW * WSTRIDE];   // double-buffered tap weights
    __shared__ float biasS[NLAY * EMB];
    __shared__ int   stateS[NROW];
    __shared__ int   pos3;
    __shared__ float meanS[NPOS], rstdS[NPOS];
    __shared__ float pooled[EMB];

    const int tid  = threadIdx.x;
    const int b    = blockIdx.x;
    const int lane = tid & 31;
    const int wv   = tid >> 5;

#if HAVE_ASYNC
    // issue stage-0 weight fetch: 2048 chunks of 16B, padded LDS rows
    {
        const bf16_t* g0 = wT;
        #pragma unroll
        for (int k = 0; k < 8; ++k) {
            int chunk = tid + k * 256;
            int row = chunk >> 4, c16 = chunk & 15;
            async_cp16((void*)(&wS[0][row * WSTRIDE + c16 * 8]),
                       (const void*)(g0 + row * EMB + c16 * 8));
        }
    }
#endif

    // ---- locate the '3' marker & stage biases ----
    const int* xb = x + b * BOARD_H * BOARD_W;
    if (tid < BOARD_H * BOARD_W) { if (xb[tid] == 3) pos3 = tid; }
    for (int i = tid; i < NLAY * EMB; i += 256) biasS[i] = conv_b[i];
    __syncthreads();

    // ---- 11x11 patch states (pad value = 4) ----
    const int r0 = pos3 / BOARD_W, c0 = pos3 % BOARD_W;
    if (tid < NROW) {
        int s = 4;
        if (tid < NPOS) {
            int br = r0 - 5 + tid / 11, bc = c0 - 5 + tid % 11;
            s = (br >= 0 && br < BOARD_H && bc >= 0 && bc < BOARD_W) ? xb[br * BOARD_W + bc] : 4;
        }
        stateS[tid] = s;
    }
    __syncthreads();

    // ---- embedding -> hA (bf16); rows 121..127 kept zero (zero-row trick) ----
    for (int i = tid; i < NROW * EMB; i += 256) {
        int p = i >> 7, ch = i & 127;
        float v = (p < NPOS) ? embed_w[stateS[p] * EMB + ch] : 0.0f;
        hA[p * WSTRIDE + ch] = (bf16_t)v;
    }

    // ---- per-lane constants ----
    const int m = (wv << 4) + (lane & 15);
    int rowoff[NTAP];
    {
        int rr = m / 11, cc = m % 11;
        #pragma unroll
        for (int t = 0; t < NTAP; ++t) {
            int nr = rr + t / 3 - 1, nc = cc + t % 3 - 1;
            bool valid = (m < NPOS) && ((unsigned)nr < 11u) && ((unsigned)nc < 11u);
            rowoff[t] = (valid ? (nr * 11 + nc) : 127) * WSTRIDE;
        }
    }
    const int aSel = ((lane >> 4) & 1) * 16;
    const int bSel = ((lane >> 4) & 1) * 16;
    const int nIdx = lane & 15;
    const int bRowBase = nIdx * WSTRIDE + bSel;   // + n*16*WSTRIDE + kc*32 (immediates)

    bf16_t* hin = hA;
    bf16_t* hout = hB;

    for (int l = 0; l < NLAY; ++l) {
        v8f acc[8];
        #pragma unroll
        for (int n = 0; n < 8; ++n) {
            v8f z = {0.f, 0.f, 0.f, 0.f, 0.f, 0.f, 0.f, 0.f};
            acc[n] = z;
        }

        #pragma unroll
        for (int t = 0; t < NTAP; ++t) {
            const int s = l * NTAP + t;
            const bf16_t* wcur = wS[s & 1];

#if HAVE_ASYNC
            wait_async0();        // my stage-s chunks are in LDS
            __syncthreads();      // everyone's are; nobody reads buffer (s+1)&1 anymore
            if (s + 1 < NSTAGE) {
                const bf16_t* gn = wT + (s + 1) * EMB * EMB;
                bf16_t* dn = wS[(s + 1) & 1];
                #pragma unroll
                for (int k = 0; k < 8; ++k) {
                    int chunk = tid + k * 256;
                    int row = chunk >> 4, c16 = chunk & 15;
                    async_cp16((void*)(dn + row * WSTRIDE + c16 * 8),
                               (const void*)(gn + row * EMB + c16 * 8));
                }
            }
#else
            __syncthreads();
            {
                const bf16_t* gs = wT + s * EMB * EMB;
                bf16_t* ds = (bf16_t*)wcur;
                for (int i = tid; i < (EMB * EMB) / 8; i += 256) {
                    int row = i >> 4, c16 = i & 15;
                    *(uint4*)(ds + row * WSTRIDE + c16 * 8) =
                        *(const uint4*)(gs + row * EMB + c16 * 8);
                }
            }
            __syncthreads();
#endif
            // ---- software-pipelined inner GEMM: 3-deep B pipeline, 2-deep A ----
            const bf16_t* wb = wcur + bRowBase;          // + n*16*WSTRIDE + kc*32
            const bf16_t* ha = hin + rowoff[t] + aSel;   // + kc*32 (A frag: +0 / +16)

            v16bf a_cur = ld_fragA(ha);
            v16bf b0 = ld_frag32(wb);                          // B[0][0]
            v16bf b1 = ld_frag32(wb + 16 * WSTRIDE);           // B[0][1]
            v16bf a_nxt = a_cur;

            #pragma unroll
            for (int kc = 0; kc < 4; ++kc) {
                if (kc < 3) a_nxt = ld_fragA(ha + (kc + 1) * 32);
                #pragma unroll
                for (int n = 0; n < 8; ++n) {
                    const int ps = kc * 8 + n + 2;      // prefetch step (distance 2)
                    v16bf bpre = b1;
                    if (ps < 32) {
                        const int pkc = ps >> 3, pn = ps & 7;
                        bpre = ld_frag32(wb + pn * 16 * WSTRIDE + pkc * 32);
                    }
                    acc[n] = __builtin_amdgcn_wmma_f32_16x16x32_bf16(
                        false, a_cur, false, b0, (short)0, acc[n], false, false);
                    b0 = b1;
                    b1 = bpre;
                }
                a_cur = a_nxt;
            }
        }

        // ---- fused bias + ReLU + residual -> hout ----
        #pragma unroll
        for (int n = 0; n < 8; ++n) {
            const int co = n * 16 + nIdx;
            const float bs = biasS[l * EMB + co];
            #pragma unroll
            for (int v = 0; v < 8; ++v) {
                int p = (wv << 4) + v + ((lane >> 4) << 3);
                float y = acc[n][v] + bs;
                y = y > 0.f ? y : 0.f;
                float hn = 0.f;
                if (p < NPOS) hn = (float)hin[p * WSTRIDE + co] + y;
                hout[p * WSTRIDE + co] = (bf16_t)hn;
            }
        }
        bf16_t* tmp = hin; hin = hout; hout = tmp;
    }
    __syncthreads();

    // ---- LayerNorm stats per position ----
    if (tid < NPOS) {
        float s = 0.f, s2 = 0.f;
        for (int ch = 0; ch < EMB; ++ch) {
            float v = (float)hin[tid * WSTRIDE + ch];
            s += v; s2 += v * v;
        }
        float mu  = s * (1.0f / EMB);
        float var = s2 * (1.0f / EMB) - mu * mu;
        meanS[tid] = mu;
        rstdS[tid] = rsqrtf(var + 1e-5f);
    }
    __syncthreads();

    // ---- normalize + mean-pool over spatial ----
    if (tid < EMB) {
        float s = 0.f;
        for (int p = 0; p < NPOS; ++p)
            s += ((float)hin[p * WSTRIDE + tid] - meanS[p]) * rstdS[p];
        pooled[tid] = (s * (1.0f / NPOS)) * ln_scale[tid] + ln_bias[tid];
    }
    __syncthreads();

    // ---- head ----
    if (tid < 4) {
        float o = head_b[tid];
        for (int c = 0; c < EMB; ++c) o += pooled[c] * head_w[c * 4 + tid];
        out[b * 4 + tid] = o;
    }
}

extern "C" void kernel_launch(void* const* d_in, const int* in_sizes, int n_in,
                              void* d_out, int out_size, void* d_ws, size_t ws_size,
                              hipStream_t stream) {
    const int*   x        = (const int*)d_in[0];
    const float* embed_w  = (const float*)d_in[1];
    const float* conv_w   = (const float*)d_in[2];
    const float* conv_b   = (const float*)d_in[3];
    const float* ln_scale = (const float*)d_in[4];
    const float* ln_bias  = (const float*)d_in[5];
    const float* head_w   = (const float*)d_in[6];
    const float* head_b   = (const float*)d_in[7];
    float* out = (float*)d_out;

    bf16_t* wT = (bf16_t*)d_ws;   // 45*128*128*2 = 1,474,560 bytes

    const int n_w = NLAY * NTAP * EMB * EMB;
    Fraser_prep_weights<<<(n_w + 255) / 256, 256, 0, stream>>>(conv_w, wT);

    const int B = in_sizes[0] / (BOARD_H * BOARD_W);
    Fraser_tetris_net<<<B, 256, 0, stream>>>(x, embed_w, wT, conv_b,
                                             ln_scale, ln_bias, head_w, head_b, out);
}
